// Net_62242666053962
// MI455X (gfx1250) — compile-verified, matrix operand order checked by reading, not run on hardware
//
#include <hip/hip_runtime.h>
#include <hip/hip_bf16.h>
#include <math.h>

#define NN 100000
#define EPSF 1e-5f

typedef __attribute__((ext_vector_type(16))) _Float16 v16h;
typedef __attribute__((ext_vector_type(8)))  float    v8f;

// ---------------- degree / normalization ----------------

__global__ void deg_init_kernel(float* deg, int n) {
    int i = blockIdx.x * blockDim.x + threadIdx.x;
    if (i < n) deg[i] = 1.0f;   // self-loop contributes 1 to every node degree
}

__global__ void deg_edge_kernel(const int* __restrict__ ei, int E, float* deg) {
    int e = blockIdx.x * blockDim.x + threadIdx.x;
    if (e < E) atomicAdd(&deg[ei[E + e]], 1.0f);   // dst row of edge_index
}

__global__ void dinv_kernel(float* dinv, int n) {
    int i = blockIdx.x * blockDim.x + threadIdx.x;
    if (i < n) dinv[i] = rsqrtf(dinv[i]);          // deg >= 1 always (self loop)
}

// ---------------- WMMA GEMM:  C[M,Nf] = A[M,K] * W[K,Nf]  (fp32 in, f16 MACs, fp32 acc) ----
// One wave32 computes a 16x32 output tile: two 16x16 accumulators share one A fragment.
// K is a template constant (128 or 256) so the K-loop fully unrolls -> back-to-back WMMAs.
// Exact tiling: M=100000 (6250*16), Nf in {256,128,64} all divisible by 32.

template <int K>
__global__ void __launch_bounds__(128)
gemm_wmma_f16_kernel(const float* __restrict__ A, const float* __restrict__ W,
                     float* __restrict__ C, int M, int Nf) {
    const int lane   = threadIdx.x & 31;
    const int tilesN = Nf >> 5;                       // 32-wide N tiles
    const int totalTiles = (M >> 4) * tilesN;
    const int tile = blockIdx.x * (blockDim.x >> 5) + (threadIdx.x >> 5);
    if (tile >= totalTiles) return;                   // uniform per wave -> EXEC stays all-1s

    const int mT = tile / tilesN;
    const int nT = tile - mT * tilesN;
    const int lh    = (lane >= 16) ? 8 : 0;           // K sub-offset for upper half-wave
    const int mrow  = (mT << 4) + (lane & 15);        // A row this lane supplies
    const int ncol0 = (nT << 5) + (lane & 15);        // first  B/C/D column this lane owns
    const int ncol1 = ncol0 + 16;                     // second B/C/D column this lane owns
    const float* __restrict__ arow = A + (size_t)mrow * K;

    v8f acc0 = {}, acc1 = {};
#pragma unroll
    for (int k0 = 0; k0 < K; k0 += 32) {
        if (k0 + 32 < K) __builtin_prefetch(arow + k0 + 32 + lh, 0, 3);  // global_prefetch_b8
        v16h a, b0, b1;
#pragma unroll
        for (int j = 0; j < 8; ++j) {
            // A 16x32 f16 layout: elems 0..7 hold K=k0+lh+0..7, elems 8..15 hold K+16
            a[j]      = (_Float16)arow[k0 + lh + j];
            a[8 + j]  = (_Float16)arow[k0 + lh + 16 + j];
            // B 32x16: lane owns one column; same K striping as A
            b0[j]     = (_Float16)W[(size_t)(k0 + lh + j) * Nf + ncol0];
            b0[8 + j] = (_Float16)W[(size_t)(k0 + lh + 16 + j) * Nf + ncol0];
            b1[j]     = (_Float16)W[(size_t)(k0 + lh + j) * Nf + ncol1];
            b1[8 + j] = (_Float16)W[(size_t)(k0 + lh + 16 + j) * Nf + ncol1];
        }
        acc0 = __builtin_amdgcn_wmma_f32_16x16x32_f16(false, a, false, b0,
                                                      (short)0, acc0, false, false);
        acc1 = __builtin_amdgcn_wmma_f32_16x16x32_f16(false, a, false, b1,
                                                      (short)0, acc1, false, false);
    }
#pragma unroll
    for (int r = 0; r < 8; ++r) {                     // C/D: VGPR r -> row r (+8 upper half-wave)
        const int m = (mT << 4) + r + lh;
        C[(size_t)m * Nf + ncol0] = acc0[r];
        C[(size_t)m * Nf + ncol1] = acc1[r];
    }
}

// ---------------- aggregation:  out[d] += dinv[s]*dinv[d]*xw[s]  (+ self loop + bias) --------

__global__ void self_init_kernel(const float* __restrict__ xw, const float* __restrict__ dinv,
                                 const float* __restrict__ bias, float* __restrict__ out,
                                 int n, int F) {
    const int chunks = F >> 2;
    long long idx = (long long)blockIdx.x * blockDim.x + threadIdx.x;
    if (idx >= (long long)n * chunks) return;
    int i = (int)(idx / chunks);
    int c = (int)(idx - (long long)i * chunks) << 2;
    float d = dinv[i];
    float s = d * d;                                  // self-loop norm
    float4 v  = *(const float4*)(xw + (size_t)i * F + c);
    float4 bb = *(const float4*)(bias + c);
    float4 o  = { s * v.x + bb.x, s * v.y + bb.y, s * v.z + bb.z, s * v.w + bb.w };
    *(float4*)(out + (size_t)i * F + c) = o;
}

// One wave32 per edge: endpoints + norm read once, float4 gathers and fp32 memory atomics
// are lane-contiguous across the destination row.
__global__ void __launch_bounds__(256)
edge_agg_kernel(const int* __restrict__ ei, int E,
                const float* __restrict__ xw, const float* __restrict__ dinv,
                float* __restrict__ out, int F) {
    const int lane = threadIdx.x & 31;
    const int e = blockIdx.x * (blockDim.x >> 5) + (threadIdx.x >> 5);
    if (e >= E) return;
    int s = ei[e];
    int d = ei[E + e];
    float nrm = dinv[s] * dinv[d];
    const float* __restrict__ xs = xw + (size_t)s * F;
    float* __restrict__ o = out + (size_t)d * F;
    for (int c = lane << 2; c < F; c += 128) {
        float4 v = *(const float4*)(xs + c);
        atomicAdd(o + c + 0, nrm * v.x);
        atomicAdd(o + c + 1, nrm * v.y);
        atomicAdd(o + c + 2, nrm * v.z);
        atomicAdd(o + c + 3, nrm * v.w);
    }
}

// ---------------- batch norm (training mode, biased var) ----------------

__global__ void __launch_bounds__(256)
bn_stats_kernel(const float* __restrict__ h, int n, int F, float* __restrict__ stats) {
    const int f = blockIdx.x;
    float s = 0.f, q = 0.f;
    for (int r = threadIdx.x; r < n; r += blockDim.x) {
        float v = h[(size_t)r * F + f];
        s += v; q += v * v;
    }
    __shared__ float ss[256], sq[256];
    ss[threadIdx.x] = s; sq[threadIdx.x] = q;
    __syncthreads();
    for (int st = 128; st > 0; st >>= 1) {
        if ((int)threadIdx.x < st) { ss[threadIdx.x] += ss[threadIdx.x + st];
                                     sq[threadIdx.x] += sq[threadIdx.x + st]; }
        __syncthreads();
    }
    if (threadIdx.x == 0) {
        float m = ss[0] / (float)n;
        stats[f]     = m;
        stats[F + f] = sq[0] / (float)n - m * m;
    }
}

__global__ void bn_apply_kernel(float* __restrict__ h, const float* __restrict__ stats,
                                const float* __restrict__ g, const float* __restrict__ beta,
                                int n, int F, int do_sigmoid) {
    long long idx = (long long)blockIdx.x * blockDim.x + threadIdx.x;
    if (idx >= (long long)n * F) return;
    int f = (int)(idx % F);
    float m = stats[f], v = stats[F + f];
    float y = (h[idx] - m) * rsqrtf(v + EPSF) * g[f] + beta[f];
    if (do_sigmoid) y = 1.0f / (1.0f + expf(-y));
    h[idx] = y;
}

// ---------------- decoder: logits[e] = dot(z[src], z[dst]) over 64 dims, one wave32 per edge --

__global__ void __launch_bounds__(256)
decode_kernel(const int* __restrict__ pos, const int* __restrict__ neg,
              int Epos, int Eneg, const float* __restrict__ z, float* __restrict__ out) {
    const int lane = threadIdx.x & 31;
    const int wave = threadIdx.x >> 5;
    const int e = blockIdx.x * (blockDim.x >> 5) + wave;
    if (e >= Epos + Eneg) return;
    int a, b;
    if (e < Epos) { a = pos[e];          b = pos[Epos + e]; }
    else          { int e2 = e - Epos;  a = neg[e2];  b = neg[Eneg + e2]; }
    const float* za = z + (size_t)a * 64;
    const float* zb = z + (size_t)b * 64;
    float s = za[lane] * zb[lane] + za[lane + 32] * zb[lane + 32];
    for (int off = 16; off > 0; off >>= 1) s += __shfl_down(s, off, 32);
    if (lane == 0) out[e] = s;
}

// ---------------- host orchestration ----------------

static inline int cdiv(long long a, long long b) { return (int)((a + b - 1) / b); }

extern "C" void kernel_launch(void* const* d_in, const int* in_sizes, int n_in,
                              void* d_out, int out_size, void* d_ws, size_t ws_size,
                              hipStream_t stream) {
    const float* x   = (const float*)d_in[0];
    const int*   ei  = (const int*)d_in[1];
    const int*   pei = (const int*)d_in[2];
    const int*   nei = (const int*)d_in[3];
    const float* W1 = (const float*)d_in[4],  *b1 = (const float*)d_in[5];
    const float* g1 = (const float*)d_in[6],  *be1 = (const float*)d_in[7];
    const float* W2 = (const float*)d_in[8],  *b2 = (const float*)d_in[9];
    const float* g2 = (const float*)d_in[10], *be2 = (const float*)d_in[11];
    const float* W3 = (const float*)d_in[12], *b3 = (const float*)d_in[13];
    const float* g3 = (const float*)d_in[14], *be3 = (const float*)d_in[15];
    float* out = (float*)d_out;

    const int E    = in_sizes[1] / 2;
    const int Epos = in_sizes[2] / 2;
    const int Eneg = in_sizes[3] / 2;

    float* ws    = (float*)d_ws;
    float* dinv  = ws;                                // NN
    float* stats = ws + NN;                           // 512
    float* xw    = stats + 512;                       // NN*256
    float* hA    = xw + (size_t)NN * 256;             // NN*256
    float* hB    = hA + (size_t)NN * 256;             // NN*128

    // normalization coefficients (shared by all 3 layers)
    deg_init_kernel<<<cdiv(NN, 256), 256, 0, stream>>>(dinv, NN);
    deg_edge_kernel<<<cdiv(E, 256), 256, 0, stream>>>(ei, E, dinv);
    dinv_kernel<<<cdiv(NN, 256), 256, 0, stream>>>(dinv, NN);

    auto gcn_layer = [&](const float* hin, int K, int F, const float* W, const float* b,
                         const float* g, const float* beta, float* hout, int sigmoid) {
        const int tiles = (NN >> 4) * (F >> 5);       // 16x32 output tiles
        if (K == 128)
            gemm_wmma_f16_kernel<128><<<cdiv(tiles, 4), 128, 0, stream>>>(hin, W, xw, NN, F);
        else
            gemm_wmma_f16_kernel<256><<<cdiv(tiles, 4), 128, 0, stream>>>(hin, W, xw, NN, F);
        self_init_kernel<<<cdiv((long long)NN * (F >> 2), 256), 256, 0, stream>>>(xw, dinv, b, hout, NN, F);
        edge_agg_kernel<<<cdiv(E, 8), 256, 0, stream>>>(ei, E, xw, dinv, hout, F);
        bn_stats_kernel<<<F, 256, 0, stream>>>(hout, NN, F, stats);
        bn_apply_kernel<<<cdiv((long long)NN * F, 256), 256, 0, stream>>>(hout, stats, g, beta, NN, F, sigmoid);
    };

    gcn_layer(x,  128, 256, W1, b1, g1, be1, hA, 1);  // 128 -> 256, sigmoid
    gcn_layer(hA, 256, 128, W2, b2, g2, be2, hB, 1);  // 256 -> 128, sigmoid
    gcn_layer(hB, 128,  64, W3, b3, g3, be3, hA, 0);  // 128 ->  64, plain BN => z

    decode_kernel<<<cdiv(Epos + Eneg, 8), 256, 0, stream>>>(pei, nei, Epos, Eneg, hA, out);
}
